// Sg2ImModel_6459630813312
// MI455X (gfx1250) — compile-verified, matrix operand order checked by reading, not run on hardware
//
#include <hip/hip_runtime.h>
#include <hip/hip_bf16.h>

// ---------------------------------------------------------------------------
// Sg2Im forward for MI455X (gfx1250, wave32).
// All GEMM-shaped math (gconv MLPs, box head, 3x3 convs as implicit GEMM)
// runs through v_wmma_f32_16x16x32_f16: f16 A/B fragments, f32 accumulators.
// B operands are PRE-PACKED into the WMMA fragment layout (one contiguous
// 32-byte block per lane per K-step) so the inner loop is b128 loads + WMMA.
// Each wave register-blocks 4 N-tiles (16x64 of C) to amortize A loads.
// ---------------------------------------------------------------------------

typedef _Float16 half16 __attribute__((ext_vector_type(16)));
typedef float    float8 __attribute__((ext_vector_type(8)));

#define N_IMG 32
#define O_OBJS 256
#define T_TRI 1024
#define EMB 64
#define GD 128
#define GH 512
#define IMG_HW 64
#define MS 16
#define FEAT_CH 192
#define NBLK 4
#define FEAT_SIZE ((size_t)N_IMG * FEAT_CH * IMG_HW * IMG_HW)

__device__ __forceinline__ int kboff(int v, int hi) {
  return ((v & 4) << 2) + ((v & 3) << 1) + (hi << 3);
}

// ---------------------------------------------------------------------------
// WMMA GEMM: out = act(A(MxK,f16 row-major) @ B + bias). B is packed:
// Bp[((tn*ksteps + ks)*32 + lane)*16 + e]  holds lane's fragment halfs.
// M mult 16, N mult 64, K mult 32. One wave -> 16x64 C tile.
// ---------------------------------------------------------------------------
__global__ void k_gemm_wmma(const _Float16* __restrict__ A,
                            const _Float16* __restrict__ Bp,
                            const float* __restrict__ bias,
                            float* __restrict__ outF,
                            _Float16* __restrict__ outH,
                            int M, int N, int K, int doRelu) {
  int wave = blockIdx.x * (blockDim.x >> 5) + (threadIdx.x >> 5);
  int tilesNB = N >> 6;
  int tiles = (M >> 4) * tilesNB;
  if (wave >= tiles) return;                 // uniform per wave: EXEC stays full
  int tm = wave / tilesNB, tnb = wave % tilesNB;
  int lane = threadIdx.x & 31;
  int r = lane & 15, hi = lane >> 4;
  int ksteps = K >> 5;
  float8 acc[NBLK];
#pragma unroll
  for (int j = 0; j < NBLK; ++j) acc[j] = (float8){0.f,0.f,0.f,0.f,0.f,0.f,0.f,0.f};
  const _Float16* Arow = A + (size_t)(tm * 16 + r) * K;
  const _Float16* Bbase = Bp + ((size_t)tnb * NBLK * ksteps * 32 + lane) * 16;
  for (int ks = 0; ks < ksteps; ++ks) {
    __builtin_prefetch(Arow + ks * 32 + 32, 0, 1);
    half16 a;
#pragma unroll
    for (int v = 0; v < 8; ++v) {
      int kb = ks * 32 + kboff(v, hi);
      a[2*v]   = Arow[kb];
      a[2*v+1] = Arow[kb + 1];
    }
#pragma unroll
    for (int j = 0; j < NBLK; ++j) {
      half16 b = *(const half16*)(Bbase + ((size_t)(j * ksteps + ks) * 32) * 16);
      acc[j] = __builtin_amdgcn_wmma_f32_16x16x32_f16(false, a, false, b,
                                                      (short)0, acc[j], false, false);
    }
  }
#pragma unroll
  for (int j = 0; j < NBLK; ++j) {
    int col = (tnb * NBLK + j) * 16 + r;
    float bv = bias ? bias[col] : 0.f;
#pragma unroll
    for (int v = 0; v < 8; ++v) {
      int row = tm * 16 + v + (hi << 3);
      float x = acc[j][v] + bv;
      if (doRelu) x = fmaxf(x, 0.f);
      size_t o = (size_t)row * N + col;
      if (outF) outF[o] = x;
      if (outH) outH[o] = (_Float16)x;
    }
  }
}

// ---------------------------------------------------------------------------
// Implicit-GEMM 3x3 SAME conv. X is NHWC f16 (A K-pairs contiguous), weights
// pre-packed like GEMM B with k = tap*Cin + c (Cin mult of 32 -> each 32-wide
// K window sits in one tap). One wave -> 16 pixels x 64 output channels.
// ---------------------------------------------------------------------------
__global__ void k_conv3x3_wmma(const _Float16* __restrict__ X,
                               const _Float16* __restrict__ Bp,
                               const float* __restrict__ bias,
                               float* __restrict__ out,
                               int NB, int H, int W, int Cin, int Cout,
                               long outNS, long outCS, int doRelu) {
  int wave = blockIdx.x * (blockDim.x >> 5) + (threadIdx.x >> 5);
  int tilesNB = Cout >> 6;
  int HW = H * W;
  int M = NB * HW;
  int tiles = (M >> 4) * tilesNB;
  if (wave >= tiles) return;
  int tm = wave / tilesNB, tnb = wave % tilesNB;
  int lane = threadIdx.x & 31;
  int r = lane & 15, hi = lane >> 4;
  int m = tm * 16 + r;
  int n = m / HW;
  int rem = m - n * HW;
  int oy = rem / W, ox = rem - (rem / W) * W;
  int ksteps = (9 * Cin) >> 5;
  float8 acc[NBLK];
#pragma unroll
  for (int j = 0; j < NBLK; ++j) acc[j] = (float8){0.f,0.f,0.f,0.f,0.f,0.f,0.f,0.f};
  const _Float16* Bbase = Bp + ((size_t)tnb * NBLK * ksteps * 32 + lane) * 16;
  for (int ks = 0; ks < ksteps; ++ks) {
    int k0 = ks * 32;
    int tap = k0 / Cin;
    int cbase = k0 - tap * Cin;
    int iy = oy + tap / 3 - 1;
    int ix = ox + tap % 3 - 1;
    bool valid = (iy >= 0) && (iy < H) && (ix >= 0) && (ix < W);
    const _Float16* px = X + ((size_t)(n * H + iy) * W + ix) * Cin;
    half16 a;
#pragma unroll
    for (int v = 0; v < 8; ++v) {
      int c = cbase + kboff(v, hi);
      _Float16 a0 = (_Float16)0.f, a1 = (_Float16)0.f;
      if (valid) { a0 = px[c]; a1 = px[c + 1]; }
      a[2*v] = a0; a[2*v+1] = a1;
    }
#pragma unroll
    for (int j = 0; j < NBLK; ++j) {
      half16 b = *(const half16*)(Bbase + ((size_t)(j * ksteps + ks) * 32) * 16);
      acc[j] = __builtin_amdgcn_wmma_f32_16x16x32_f16(false, a, false, b,
                                                      (short)0, acc[j], false, false);
    }
  }
#pragma unroll
  for (int j = 0; j < NBLK; ++j) {
    int co = (tnb * NBLK + j) * 16 + r;
    float bv = bias[co];
#pragma unroll
    for (int v = 0; v < 8; ++v) {
      int mm = tm * 16 + v + (hi << 3);
      int nn = mm / HW;
      int rr = mm - nn * HW;
      float x = acc[j][v] + bv;
      if (doRelu) x = fmaxf(x, 0.f);
      out[(size_t)nn * outNS + (size_t)co * outCS + rr] = x;
    }
  }
}

// ------------------- weight packing into WMMA fragment layout --------------
// Bp element i: e=i&15, lane=(i>>4)&31, ks=(i>>9)%ksteps, tn=(i>>9)/ksteps.
// k = ks*32 + kboff(e>>1, lane>>4) + (e&1);  n = tn*16 + (lane&15).
__global__ void k_pack_w(const float* __restrict__ Wm, _Float16* __restrict__ Bp,
                         int K, int N) {
  long total = (long)K * N;
  long i = blockIdx.x * (long)blockDim.x + threadIdx.x;
  if (i >= total) return;
  int e = i & 15;
  int lane = (i >> 4) & 31;
  long rest = i >> 9;
  int ksteps = K >> 5;
  int ks = rest % ksteps;
  int tn = rest / ksteps;
  int k = ks * 32 + kboff(e >> 1, lane >> 4) + (e & 1);
  int col = tn * 16 + (lane & 15);
  Bp[i] = (_Float16)Wm[(size_t)k * N + col];
}
__global__ void k_pack_convw(const float* __restrict__ cw, _Float16* __restrict__ Bp,
                             int Cin, int Cout) {
  int K = 9 * Cin;
  long total = (long)K * Cout;
  long i = blockIdx.x * (long)blockDim.x + threadIdx.x;
  if (i >= total) return;
  int e = i & 15;
  int lane = (i >> 4) & 31;
  long rest = i >> 9;
  int ksteps = K >> 5;
  int ks = rest % ksteps;
  int tn = rest / ksteps;
  int k = ks * 32 + kboff(e >> 1, lane >> 4) + (e & 1);
  int co = tn * 16 + (lane & 15);
  int tap = k / Cin, c = k - tap * Cin;
  Bp[i] = (_Float16)cw[((size_t)co * Cin + c) * 9 + tap];
}

// ------------------------------- scalar helpers ----------------------------
__global__ void k_zero_f32(float* p, long n) {
  long i = blockIdx.x * (long)blockDim.x + threadIdx.x;
  if (i < n) p[i] = 0.f;
}
__global__ void k_f32_to_f16(const float* __restrict__ s, _Float16* __restrict__ d, long n) {
  long i = blockIdx.x * (long)blockDim.x + threadIdx.x;
  if (i < n) d[i] = (_Float16)s[i];
}
__global__ void k_gather(const float* __restrict__ tab, const int* __restrict__ idx,
                         int stride, int off, float* __restrict__ out, int rows, int dim) {
  long i = blockIdx.x * (long)blockDim.x + threadIdx.x;
  if (i >= (long)rows * dim) return;
  int r = i / dim, c = i - (long)r * dim;
  out[i] = tab[(size_t)idx[r * stride + off] * dim + c];
}
__global__ void k_build_t(const float* __restrict__ ov, const float* __restrict__ pv,
                          const int* __restrict__ tri, _Float16* __restrict__ t,
                          int T, int din) {
  long i = blockIdx.x * (long)blockDim.x + threadIdx.x;
  long total = (long)T * 3 * din;
  if (i >= total) return;
  int r = i / (3 * din), c = i - (long)r * 3 * din;
  float v;
  if (c < din)            v = ov[(size_t)tri[3 * r] * din + c];
  else if (c < 2 * din)   v = pv[(size_t)r * din + (c - din)];
  else                    v = ov[(size_t)tri[3 * r + 2] * din + (c - 2 * din)];
  t[i] = (_Float16)v;
}
__global__ void k_pool_atomic(const float* __restrict__ newt, const int* __restrict__ tri,
                              float* pooled, float* counts, int T) {
  long i = blockIdx.x * (long)blockDim.x + threadIdx.x;
  if (i >= (long)T * GH) return;
  int r = i >> 9, c = i & (GH - 1);
  int s = tri[3 * r], o = tri[3 * r + 2];
  atomicAdd(&pooled[(size_t)s * GH + c], newt[(size_t)r * (2 * GH + GD) + c]);
  atomicAdd(&pooled[(size_t)o * GH + c], newt[(size_t)r * (2 * GH + GD) + GH + GD + c]);
  if (c == 0) { atomicAdd(&counts[s], 1.f); atomicAdd(&counts[o], 1.f); }
}
__global__ void k_pool_div(float* pooled, const float* counts) {
  long i = blockIdx.x * (long)blockDim.x + threadIdx.x;
  if (i >= (long)O_OBJS * GH) return;
  pooled[i] /= fmaxf(counts[i >> 9], 1.f);
}
__global__ void k_copy_pred(const float* __restrict__ newt, float* __restrict__ pv, int T) {
  long i = blockIdx.x * (long)blockDim.x + threadIdx.x;
  if (i >= (long)T * GD) return;
  int r = i >> 7, c = i & (GD - 1);
  pv[i] = newt[(size_t)r * (2 * GH + GD) + GH + c];
}
__global__ void k_box_out(const float* __restrict__ h, const float* __restrict__ w2,
                          const float* __restrict__ b2, float* __restrict__ out) {
  int i = blockIdx.x * blockDim.x + threadIdx.x;
  if (i >= O_OBJS * 4) return;
  int o = i >> 2, j = i & 3;
  float s = b2[j];
  for (int k = 0; k < GH; ++k) s += h[(size_t)o * GH + k] * w2[k * 4 + j];
  out[i] = fmaxf(s, 0.f);
}
__global__ void k_upsample2(const float* __restrict__ in, float* __restrict__ out,
                            int NC, int s) {
  int os = 2 * s;
  long total = (long)NC * os * os;
  long i = blockIdx.x * (long)blockDim.x + threadIdx.x;
  if (i >= total) return;
  int xy = i % (os * os);
  long nc = i / (os * os);
  int y = xy / os, x = xy - (xy / os) * os;
  out[i] = in[nc * s * s + (y >> 1) * s + (x >> 1)];
}
__global__ void k_bn_stats(const float* __restrict__ x, float* mean, float* rinv,
                           int N, int C, int HW) {
  int c = blockIdx.x;
  __shared__ float s1[256], s2[256];
  float a = 0.f, b = 0.f;
  long cnt = (long)N * HW;
  for (long i = threadIdx.x; i < cnt; i += blockDim.x) {
    int n = i / HW, p = i - (long)(i / HW) * HW;
    float v = x[((size_t)n * C + c) * HW + p];
    a += v; b += v * v;
  }
  s1[threadIdx.x] = a; s2[threadIdx.x] = b; __syncthreads();
  for (int st = 128; st > 0; st >>= 1) {
    if ((int)threadIdx.x < st) { s1[threadIdx.x] += s1[threadIdx.x + st];
                                 s2[threadIdx.x] += s2[threadIdx.x + st]; }
    __syncthreads();
  }
  if (threadIdx.x == 0) {
    float m = s1[0] / (float)cnt;
    float v = s2[0] / (float)cnt - m * m;
    mean[c] = m; rinv[c] = rsqrtf(v + 1e-5f);
  }
}
__global__ void k_bn_apply_nhwc(const float* __restrict__ x, const float* __restrict__ mean,
                                const float* __restrict__ rinv, const float* __restrict__ g,
                                const float* __restrict__ bb, _Float16* __restrict__ out,
                                int N, int C, int HW, int W) {
  long total = (long)N * C * HW;
  long i = blockIdx.x * (long)blockDim.x + threadIdx.x;
  if (i >= total) return;
  int p = i % HW;
  long nc = i / HW;
  int c = nc % C, n = nc / C;
  int y = p / W, xw = p - (p / W) * W;
  float v = g[c] * (x[i] - mean[c]) * rinv[c] + bb[c];
  out[(((size_t)n * (HW / W) + y) * W + xw) * C + c] = (_Float16)v;
}
__global__ void k_mask_out(const float* __restrict__ x, const float* __restrict__ w,
                           const float* __restrict__ b, float* __restrict__ masks) {
  int i = blockIdx.x * blockDim.x + threadIdx.x;
  if (i >= O_OBJS * MS * MS) return;
  int n = i >> 8, p = i & 255;
  float s = b[0];
  for (int c = 0; c < GD; ++c) s += x[((size_t)n * GD + c) * (MS * MS) + p] * w[c];
  masks[i] = 1.f / (1.f + expf(-s));
}
__global__ void k_sample_mask(const float* __restrict__ masks, const float* __restrict__ boxes,
                              float* __restrict__ smask) {
  long i = blockIdx.x * (long)blockDim.x + threadIdx.x;
  if (i >= (long)O_OBJS * IMG_HW * IMG_HW) return;
  int n = i >> 12, p = i & 4095;
  int y = p >> 6, x = p & 63;
  float x0 = boxes[n * 4 + 0], y0 = boxes[n * 4 + 1];
  float x1 = boxes[n * 4 + 2], y1 = boxes[n * 4 + 3];
  float Xn = ((float)x / 63.f - x0) / (x1 - x0);
  float Yn = ((float)y / 63.f - y0) / (y1 - y0);
  float ix = Xn * 16.f - 0.5f, iy = Yn * 16.f - 0.5f;
  float fx0 = floorf(ix), fy0 = floorf(iy);
  float wx1 = ix - fx0, wy1 = iy - fy0;
  int x0i = (int)fx0, y0i = (int)fy0;
  float acc = 0.f;
  for (int dy = 0; dy < 2; ++dy)
    for (int dx = 0; dx < 2; ++dx) {
      int xi = x0i + dx, yi = y0i + dy;
      float w = (dx ? wx1 : 1.f - wx1) * (dy ? wy1 : 1.f - wy1);
      bool valid = (xi >= 0) && (xi < MS) && (yi >= 0) && (yi < MS);
      int xc = min(max(xi, 0), MS - 1), yc = min(max(yi, 0), MS - 1);
      acc += masks[n * 256 + yc * MS + xc] * w * (valid ? 1.f : 0.f);
    }
  smask[i] = acc;
}
__global__ void k_starts(const int* __restrict__ o2i, int* __restrict__ starts) {
  int i = threadIdx.x;
  if (i > N_IMG) return;
  int cnt = 0;
  for (int n = 0; n < O_OBJS; ++n) cnt += (o2i[n] < i) ? 1 : 0;
  starts[i] = cnt;
}
__global__ void k_layout(const float* __restrict__ ov, const float* __restrict__ smask,
                         const int* __restrict__ starts, float* __restrict__ out) {
  long i = blockIdx.x * (long)blockDim.x + threadIdx.x;
  if (i >= (long)N_IMG * GD * IMG_HW * IMG_HW) return;
  int p = i & 4095;
  int c = (i >> 12) & (GD - 1);
  int img = i >> 19;
  float s = 0.f;
  int e = starts[img + 1];
  for (int n = starts[img]; n < e; ++n)
    s += ov[(size_t)n * GD + c] * smask[(size_t)n * 4096 + p];
  out[((size_t)img * FEAT_CH + c) * 4096 + p] = s;
}
__global__ void k_conv1_direct(const float* __restrict__ img, const float* __restrict__ w,
                               const float* __restrict__ b, float* __restrict__ out) {
  long i = blockIdx.x * (long)blockDim.x + threadIdx.x;
  if (i >= (long)N_IMG * 32 * 4096) return;
  int p = i & 4095, y = p >> 6, x = p & 63;
  int co = (i >> 12) & 31, n = i >> 17;
  float s = b[co];
  for (int ci = 0; ci < 3; ++ci)
    for (int ky = 0; ky < 3; ++ky)
      for (int kx = 0; kx < 3; ++kx) {
        int iy = y + ky - 1, ix = x + kx - 1;
        if (iy >= 0 && iy < 64 && ix >= 0 && ix < 64)
          s += img[((size_t)n * 3 + ci) * 4096 + iy * 64 + ix] *
               w[((co * 3 + ci) * 3 + ky) * 3 + kx];
      }
  out[i] = s;
}
__global__ void k_in_stats(const float* __restrict__ x, float* mean, float* rinv) {
  int nc = blockIdx.x;
  __shared__ float s1[256], s2[256];
  const float* base = x + (size_t)nc * 4096;
  float a = 0.f, b = 0.f;
  for (int i = threadIdx.x; i < 4096; i += blockDim.x) {
    float v = base[i]; a += v; b += v * v;
  }
  s1[threadIdx.x] = a; s2[threadIdx.x] = b; __syncthreads();
  for (int st = 128; st > 0; st >>= 1) {
    if ((int)threadIdx.x < st) { s1[threadIdx.x] += s1[threadIdx.x + st];
                                 s2[threadIdx.x] += s2[threadIdx.x + st]; }
    __syncthreads();
  }
  if (threadIdx.x == 0) {
    float m = s1[0] / 4096.f;
    float v = s2[0] / 4096.f - m * m;
    mean[nc] = m; rinv[nc] = rsqrtf(v + 1e-5f);
  }
}
__global__ void k_in_apply_nhwc(const float* __restrict__ x, const float* __restrict__ mean,
                                const float* __restrict__ rinv, _Float16* __restrict__ out) {
  long i = blockIdx.x * (long)blockDim.x + threadIdx.x;
  if (i >= (long)N_IMG * 32 * 4096) return;
  int nc = i >> 12;
  int n = nc >> 5, c = nc & 31;
  int p = i & 4095, y = p >> 6, xw = p & 63;
  float v = (x[i] - mean[nc]) * rinv[nc];
  v = v > 0.f ? v : 0.2f * v;
  out[(((size_t)n * 64 + y) * 64 + xw) * 32 + c] = (_Float16)v;
}

// ---------------------------------------------------------------------------
extern "C" void kernel_launch(void* const* d_in, const int* in_sizes, int n_in,
                              void* d_out, int out_size, void* d_ws, size_t ws_size,
                              hipStream_t stream) {
  (void)in_sizes; (void)n_in; (void)out_size; (void)ws_size;
  auto F = [&](int i) { return (const float*)d_in[i]; };
  auto Ii = [&](int i) { return (const int*)d_in[i]; };

  // setup_inputs() / make_params() insertion order:
  const float* img      = F(0);
  const int*   objs     = Ii(1);
  const int*   triples  = Ii(2);
  const int*   o2i      = Ii(3);
  const float* boxes_gt = F(4);
  const float* obj_emb  = F(5);
  const float* pred_emb = F(6);
  const int GC = 7;                 // gconvs[l]: w1,b1,w2,b2,w3,b3,w4,b4
  const int BX = GC + 5 * 8;        // 47: box w1,b1,w2,b2
  const int MB = BX + 4;            // 51: mask_blocks[i]: g,b,cw,cb
  const int MOW = MB + 16, MOB = MOW + 1;       // 67,68
  const int C1W = MOB + 1, C1B = C1W + 1;       // 69,70
  const int C2W = C1B + 1, C2B = C2W + 1;       // 71,72

  // workspace carve-out
  char* ws = (char*)d_ws;
  size_t off = 0;
  auto alloc = [&](size_t bytes) -> void* {
    off = (off + 255) & ~(size_t)255;
    void* p = ws + off; off += bytes; return p;
  };
  float*    obj_vecs = (float*)alloc(O_OBJS * GD * 4);
  float*    pred_vec = (float*)alloc(T_TRI * GD * 4);
  _Float16* t16      = (_Float16*)alloc((size_t)T_TRI * 3 * GD * 2);
  _Float16* h16      = (_Float16*)alloc((size_t)T_TRI * GH * 2);
  float*    newt     = (float*)alloc((size_t)T_TRI * (2 * GH + GD) * 4);
  float*    pooled   = (float*)alloc(O_OBJS * GH * 4);
  float*    counts   = (float*)alloc(O_OBJS * 4);
  _Float16* pooled16 = (_Float16*)alloc(O_OBJS * GH * 2);
  _Float16* bp16     = (_Float16*)alloc((size_t)GH * (2 * GH + GD) * 2); // packed B
  _Float16* h2_16    = (_Float16*)alloc(O_OBJS * GH * 2);
  _Float16* obj16    = (_Float16*)alloc(O_OBJS * GD * 2);
  float*    boxh32   = (float*)alloc(O_OBJS * GH * 4);
  float*    bufA     = (float*)alloc((size_t)O_OBJS * GD * MS * MS * 4);
  float*    bufB     = (float*)alloc((size_t)O_OBJS * GD * MS * MS * 4);
  _Float16* nhwc16   = (_Float16*)alloc((size_t)O_OBJS * MS * MS * GD * 2);
  float*    bnm      = (float*)alloc(GD * 4);
  float*    bni      = (float*)alloc(GD * 4);
  float*    masks    = (float*)alloc(O_OBJS * MS * MS * 4);
  float*    smask    = (float*)alloc((size_t)O_OBJS * IMG_HW * IMG_HW * 4);
  int*      starts   = (int*)alloc((N_IMG + 1) * 4);
  float*    inm      = (float*)alloc(N_IMG * 32 * 4);
  float*    inr      = (float*)alloc(N_IMG * 32 * 4);
  float*    y1       = bufB;                     // reuse after mask stage
  _Float16* y1h      = nhwc16;                   // reuse after mask stage
  float*    dout     = (float*)d_out;

#define L1(kern, n, ...) kern<<<(int)(((long)(n) + 255) / 256), 256, 0, stream>>>(__VA_ARGS__)
  auto gemm = [&](const _Float16* A, const _Float16* B, const float* bias,
                  float* oF, _Float16* oH, int M, int N, int K, int relu) {
    int tiles = (M / 16) * (N / 64);
    k_gemm_wmma<<<(tiles + 7) / 8, 256, 0, stream>>>(A, B, bias, oF, oH, M, N, K, relu);
  };
  auto conv = [&](const _Float16* X, const _Float16* Wt, const float* bias, float* out,
                  int NB, int H, int W, int Cin, int Cout, long oNS, long oCS, int relu) {
    int tiles = (NB * H * W / 16) * (Cout / 64);
    k_conv3x3_wmma<<<(tiles + 7) / 8, 256, 0, stream>>>(X, Wt, bias, out, NB, H, W,
                                                        Cin, Cout, oNS, oCS, relu);
  };

  // ---- embeddings ----
  L1(k_gather, O_OBJS * EMB, obj_emb, objs, 1, 0, obj_vecs, O_OBJS, EMB);
  L1(k_gather, T_TRI * EMB, pred_emb, triples, 3, 1, pred_vec, T_TRI, EMB);

  // ---- 5 graph-conv layers ----
  for (int l = 0; l < 5; ++l) {
    int din = (l == 0) ? EMB : GD;
    int g = GC + 8 * l;
    L1(k_build_t, (long)T_TRI * 3 * din, obj_vecs, pred_vec, triples, t16, T_TRI, din);
    L1(k_pack_w, (long)3 * din * GH, F(g + 0), bp16, 3 * din, GH);
    gemm(t16, bp16, F(g + 1), nullptr, h16, T_TRI, GH, 3 * din, 1);
    L1(k_pack_w, (long)GH * (2 * GH + GD), F(g + 2), bp16, GH, 2 * GH + GD);
    gemm(h16, bp16, F(g + 3), newt, nullptr, T_TRI, 2 * GH + GD, GH, 1);
    L1(k_zero_f32, O_OBJS * GH, pooled, (long)O_OBJS * GH);
    L1(k_zero_f32, O_OBJS, counts, (long)O_OBJS);
    L1(k_pool_atomic, (long)T_TRI * GH, newt, triples, pooled, counts, T_TRI);
    L1(k_pool_div, O_OBJS * GH, pooled, counts);
    L1(k_copy_pred, T_TRI * GD, newt, pred_vec, T_TRI);
    L1(k_f32_to_f16, O_OBJS * GH, pooled, pooled16, (long)O_OBJS * GH);
    L1(k_pack_w, (long)GH * GH, F(g + 4), bp16, GH, GH);
    gemm(pooled16, bp16, F(g + 5), nullptr, h2_16, O_OBJS, GH, GH, 1);
    L1(k_pack_w, (long)GH * GD, F(g + 6), bp16, GH, GD);
    gemm(h2_16, bp16, F(g + 7), obj_vecs, obj16, O_OBJS, GD, GH, 1);
  }

  // ---- box head ----
  L1(k_pack_w, (long)GD * GH, F(BX + 0), bp16, GD, GH);
  gemm(obj16, bp16, F(BX + 1), boxh32, nullptr, O_OBJS, GH, GD, 1);
  L1(k_box_out, O_OBJS * 4, boxh32, F(BX + 2), F(BX + 3), dout + FEAT_SIZE);

  // ---- mask decoder: 4 x (upsample2, BN, 3x3 conv + relu) ----
  const float* cur = obj_vecs;  // (256,128,1,1) NCHW
  int s = 1;
  for (int blk = 0; blk < 4; ++blk) {
    int os = 2 * s, hw = os * os;
    int b = MB + 4 * blk;  // g,b,cw,cb
    L1(k_upsample2, (long)O_OBJS * GD * hw, cur, bufB, O_OBJS * GD, s);
    k_bn_stats<<<GD, 256, 0, stream>>>(bufB, bnm, bni, O_OBJS, GD, hw);
    L1(k_bn_apply_nhwc, (long)O_OBJS * GD * hw, bufB, bnm, bni, F(b + 0), F(b + 1),
       nhwc16, O_OBJS, GD, hw, os);
    L1(k_pack_convw, (long)GD * GD * 9, F(b + 2), bp16, GD, GD);
    conv(nhwc16, bp16, F(b + 3), bufA, O_OBJS, os, os, GD, GD,
         (long)GD * hw, (long)hw, 1);
    cur = bufA; s = os;
  }
  L1(k_mask_out, O_OBJS * MS * MS, bufA, F(MOW), F(MOB), masks);

  // ---- layout branch (channels 0..127 of features) ----
  L1(k_sample_mask, (long)O_OBJS * IMG_HW * IMG_HW, masks, boxes_gt, smask);
  k_starts<<<1, 64, 0, stream>>>(o2i, starts);
  L1(k_layout, (long)N_IMG * GD * IMG_HW * IMG_HW, obj_vecs, smask, starts, dout);

  // ---- image branch (channels 128..191 of features) ----
  L1(k_conv1_direct, (long)N_IMG * 32 * 4096, img, F(C1W), F(C1B), y1);
  k_in_stats<<<N_IMG * 32, 256, 0, stream>>>(y1, inm, inr);
  L1(k_in_apply_nhwc, (long)N_IMG * 32 * 4096, y1, inm, inr, y1h);
  L1(k_pack_convw, (long)64 * 32 * 9, F(C2W), bp16, 32, 64);
  conv(y1h, bp16, F(C2B), dout + (size_t)GD * 4096, N_IMG, IMG_HW, IMG_HW, 32, 64,
       (long)FEAT_CH * 4096, 4096L, 0);
#undef L1
}
